// TiledGEMM_57784490000873
// MI455X (gfx1250) — matrix operand order repaired and checked by reading, not compiled, not run on hardware
//
#include <hip/hip_runtime.h>

// CDNA5 / gfx1250 block-scaled (MX) dequant fused GEMM, bf16x3 fp32 emulation.
//
// Phase 1 (prep): fold per-32-block scales into a (hi, lo) bf16 split of both
//   operands, stored in workspace. One streaming pass, ~400MB traffic (~17us).
// Phase 2 (GEMM): pure-bf16 tiles -> LDS (double-buffered, async-to-LDS when
//   available) -> v_wmma_f32_16x16x32_bf16. acc += Ahi*Bhi + Alo*Bhi + Ahi*Blo.
// Fallback: if workspace too small, run the fused single-kernel variant.

typedef __attribute__((ext_vector_type(16))) __bf16 v16bf;
typedef __attribute__((ext_vector_type(8)))  __bf16 v8bf;
typedef __attribute__((ext_vector_type(4)))  __bf16 v4bf;
typedef __attribute__((ext_vector_type(8)))  float  v8f;
typedef __attribute__((ext_vector_type(4)))  int    v4i;

#define TM 128           // block tile rows (M)
#define TN 128           // block tile cols (N)
#define TK 32            // K step == MX block size == bf16 WMMA K
#define LDT 40           // padded LDS row stride in bf16 elems (80B, 16B-aligned, conflict-free)

#if defined(__has_builtin)
#if __has_builtin(__builtin_amdgcn_global_load_async_to_lds_b128)
#define MX_ASYNC 1
#endif
#endif

typedef __attribute__((address_space(1))) v4i as1_v4i;
typedef __attribute__((address_space(3))) v4i as3_v4i;

// ---------------------------------------------------------------- helpers ---
__device__ __forceinline__ v16bf cat8(v8bf a, v8bf b) {
    return __builtin_shufflevector(a, b, 0, 1, 2, 3, 4, 5, 6, 7,
                                         8, 9, 10, 11, 12, 13, 14, 15);
}

__device__ __forceinline__ void split4(float4 f, float s, v4bf& h, v4bf& l) {
    float v0 = f.x * s, v1 = f.y * s, v2 = f.z * s, v3 = f.w * s;
    h[0] = (__bf16)v0; l[0] = (__bf16)(v0 - (float)h[0]);
    h[1] = (__bf16)v1; l[1] = (__bf16)(v1 - (float)h[1]);
    h[2] = (__bf16)v2; l[2] = (__bf16)(v2 - (float)h[2]);
    h[3] = (__bf16)v3; l[3] = (__bf16)(v3 - (float)h[3]);
}

__device__ __forceinline__ void async_cp16(const __bf16* g, __bf16* l) {
#ifdef MX_ASYNC
    __builtin_amdgcn_global_load_async_to_lds_b128(
        (as1_v4i*)g, (as3_v4i*)l, 0, 0);
#else
    *(v8bf*)l = *(const v8bf*)g;   // global_load_b128 + ds_store_b128
#endif
}

// ------------------------------------------------------ phase 1: MX split ---
__global__ void __launch_bounds__(256)
mx_split_kernel(const float* __restrict__ src, const float* __restrict__ scl,
                __bf16* __restrict__ hi, __bf16* __restrict__ lo,
                int C, int nblk, long total4) {
    long idx = (long)blockIdx.x * 256 + threadIdx.x;
    if (idx >= total4) return;
    long flat = idx * 4;
    long row  = flat / C;
    int  c    = (int)(flat - row * C);
    float s   = scl[row * nblk + (c >> 5)];
    float4 f  = *(const float4*)&src[flat];
    v4bf h, l;
    split4(f, s, h, l);
    *(v4bf*)&hi[flat] = h;
    *(v4bf*)&lo[flat] = l;
}

// ---------------------------------------------- phase 2: bf16x3 WMMA GEMM ---
__global__ void __launch_bounds__(256)
mxfp_gemm_v2_kernel(const __bf16* __restrict__ AHi, const __bf16* __restrict__ ALo,
                    const __bf16* __restrict__ BHi, const __bf16* __restrict__ BLo,
                    const float* __restrict__ bias, float* __restrict__ out,
                    int M, int N, int C) {
    extern __shared__ __attribute__((aligned(16))) __bf16 smem[];
    const int nblk = C >> 5;
    constexpr int PLANE = TM * LDT;     // 5120 bf16
    constexpr int BUF   = 4 * PLANE;    // Ahi/Alo/Bhi/Blo = 40KB

    const int tid   = threadIdx.x;
    const int lane  = tid & 31;
    const int wave  = tid >> 5;
    const int waveM = wave >> 2;        // 0..1
    const int waveN = wave & 3;         // 0..3
    const int l16   = lane & 15;
    const int half  = lane >> 4;

    const int tileN0 = blockIdx.x * TN; // x innermost: weight panel L2-resident
    const int tileM0 = blockIdx.y * TM;

    // 16B-chunk staging assignment: tile = 128 rows x 4 chunks of 8 bf16.
    int cRow[2], cCh[2];
#pragma unroll
    for (int i = 0; i < 2; ++i) {
        int idx = tid + i * 256;
        cRow[i] = idx >> 2;             // 0..127
        cCh[i]  = (idx & 3) << 3;       // 0,8,16,24
    }

    auto stage = [&](int kb, int b) {
        const int k0 = kb * TK;
        __bf16* base = smem + b * BUF;
#pragma unroll
        for (int i = 0; i < 2; ++i) {
            const size_t ga = (size_t)(tileM0 + cRow[i]) * C + k0 + cCh[i];
            const size_t gb = (size_t)(tileN0 + cRow[i]) * C + k0 + cCh[i];
            const int    ld = cRow[i] * LDT + cCh[i];
            async_cp16(AHi + ga, base + ld);
            async_cp16(ALo + ga, base + PLANE + ld);
            async_cp16(BHi + gb, base + 2 * PLANE + ld);
            async_cp16(BLo + gb, base + 3 * PLANE + ld);
        }
    };

    v8f acc[4][2] = {};
    stage(0, 0);

    for (int kb = 0; kb < nblk; ++kb) {
        const int b = kb & 1;
#ifdef MX_ASYNC
        asm volatile("s_wait_asynccnt 0x0" ::: "memory");
#endif
        __syncthreads();                     // staged tiles visible to all waves
        if (kb + 1 < nblk) stage(kb + 1, b ^ 1);   // overlap next copy with WMMA

        const __bf16* pAhi = smem + b * BUF;
        const __bf16* pAlo = pAhi + PLANE;
        const __bf16* pBhi = pAhi + 2 * PLANE;
        const __bf16* pBlo = pAhi + 3 * PLANE;

        // B fragments: 32x16 bf16, lane half selects K-octets {0..7,16..23}/{8..15,24..31}
        v16bf bh[2], bl[2];
#pragma unroll
        for (int ni = 0; ni < 2; ++ni) {
            const int col  = waveN * 32 + ni * 16 + l16;
            const int base = col * LDT + half * 8;
            bh[ni] = cat8(*(const v8bf*)&pBhi[base], *(const v8bf*)&pBhi[base + 16]);
            bl[ni] = cat8(*(const v8bf*)&pBlo[base], *(const v8bf*)&pBlo[base + 16]);
        }

        // A fragments: 16x32 bf16, lane half selects K-octets {0..7,16..23}/{8..15,24..31}
#pragma unroll
        for (int mi = 0; mi < 4; ++mi) {
            const int row  = waveM * 64 + mi * 16 + l16;
            const int base = row * LDT + half * 8;
            v16bf ah = cat8(*(const v8bf*)&pAhi[base], *(const v8bf*)&pAhi[base + 16]);
            v16bf al = cat8(*(const v8bf*)&pAlo[base], *(const v8bf*)&pAlo[base + 16]);
#pragma unroll
            for (int ni = 0; ni < 2; ++ni) {
                acc[mi][ni] = __builtin_amdgcn_wmma_f32_16x16x32_bf16(
                    false, ah, false, bh[ni], (short)0, acc[mi][ni], false, false);
                acc[mi][ni] = __builtin_amdgcn_wmma_f32_16x16x32_bf16(
                    false, al, false, bh[ni], (short)0, acc[mi][ni], false, false);
                acc[mi][ni] = __builtin_amdgcn_wmma_f32_16x16x32_bf16(
                    false, ah, false, bl[ni], (short)0, acc[mi][ni], false, false);
            }
        }
        __syncthreads();                     // reads done before buffer reuse
    }

    // Epilogue: C/D layout = 8 VGPRs, vgpr r -> row r + 8*lanehalf, col = lane%16
#pragma unroll
    for (int ni = 0; ni < 2; ++ni) {
        const int col = tileN0 + waveN * 32 + ni * 16 + l16;
        const float bcol = bias[col];
#pragma unroll
        for (int mi = 0; mi < 4; ++mi) {
            const int row0 = tileM0 + waveM * 64 + mi * 16 + half * 8;
#pragma unroll
            for (int r = 0; r < 8; ++r) {
                out[(size_t)(row0 + r) * N + col] = acc[mi][ni][r] + bcol;
            }
        }
    }
}

// ----------------------------------- fallback: fused single-kernel variant ---
__device__ __forceinline__ void cvt_store4(__bf16* __restrict__ hi,
                                           __bf16* __restrict__ lo,
                                           int off, float4 f, float s) {
    v4bf h, l;
    split4(f, s, h, l);
    *(v4bf*)&hi[off] = h;
    *(v4bf*)&lo[off] = l;
}

__global__ void __launch_bounds__(256)
mxfp_gemm_fused_kernel(const float* __restrict__ X, const float* __restrict__ W,
                       const float* __restrict__ bias, const float* __restrict__ SX,
                       const float* __restrict__ SW, float* __restrict__ out,
                       int M, int N, int C) {
    const int nblk = C >> 5;
    __shared__ alignas(16) __bf16 sAhi[TM * LDT];
    __shared__ alignas(16) __bf16 sAlo[TM * LDT];
    __shared__ alignas(16) __bf16 sBhi[TN * LDT];
    __shared__ alignas(16) __bf16 sBlo[TN * LDT];

    const int tid = threadIdx.x, lane = tid & 31, wave = tid >> 5;
    const int waveM = wave >> 2, waveN = wave & 3;
    const int l16 = lane & 15, half = lane >> 4;
    const int tileN0 = blockIdx.x * TN, tileM0 = blockIdx.y * TM;

    v8f acc[4][2] = {};
    float4 aR[4], bR[4];
    float  aS[4], bS[4];
    int rRow[4], rCol[4];
#pragma unroll
    for (int i = 0; i < 4; ++i) {
        int idx = tid + i * 256;
        rRow[i] = idx >> 3;
        rCol[i] = (idx & 7) << 2;
    }
    auto load_tiles = [&](int kb) {
        const int k0 = kb * TK;
#pragma unroll
        for (int i = 0; i < 4; ++i) {
            const size_t ar = (size_t)(tileM0 + rRow[i]);
            const size_t br = (size_t)(tileN0 + rRow[i]);
            aR[i] = *(const float4*)&X[ar * C + k0 + rCol[i]];
            bR[i] = *(const float4*)&W[br * C + k0 + rCol[i]];
            aS[i] = SX[ar * nblk + kb];
            bS[i] = SW[br * nblk + kb];
        }
    };
    load_tiles(0);
    for (int kb = 0; kb < nblk; ++kb) {
        __syncthreads();
#pragma unroll
        for (int i = 0; i < 4; ++i) {
            const int off = rRow[i] * LDT + rCol[i];
            cvt_store4(sAhi, sAlo, off, aR[i], aS[i]);
            cvt_store4(sBhi, sBlo, off, bR[i], bS[i]);
        }
        __syncthreads();
        if (kb + 1 < nblk) load_tiles(kb + 1);

        v16bf bh[2], bl[2];
#pragma unroll
        for (int ni = 0; ni < 2; ++ni) {
            const int col = waveN * 32 + ni * 16 + l16;
            const int base = col * LDT + half * 8;
            bh[ni] = cat8(*(const v8bf*)&sBhi[base], *(const v8bf*)&sBhi[base + 16]);
            bl[ni] = cat8(*(const v8bf*)&sBlo[base], *(const v8bf*)&sBlo[base + 16]);
        }
#pragma unroll
        for (int mi = 0; mi < 4; ++mi) {
            const int row = waveM * 64 + mi * 16 + l16;
            const int base = row * LDT + half * 8;
            v16bf ah = cat8(*(const v8bf*)&sAhi[base], *(const v8bf*)&sAhi[base + 16]);
            v16bf al = cat8(*(const v8bf*)&sAlo[base], *(const v8bf*)&sAlo[base + 16]);
#pragma unroll
            for (int ni = 0; ni < 2; ++ni) {
                acc[mi][ni] = __builtin_amdgcn_wmma_f32_16x16x32_bf16(
                    false, ah, false, bh[ni], (short)0, acc[mi][ni], false, false);
                acc[mi][ni] = __builtin_amdgcn_wmma_f32_16x16x32_bf16(
                    false, al, false, bh[ni], (short)0, acc[mi][ni], false, false);
                acc[mi][ni] = __builtin_amdgcn_wmma_f32_16x16x32_bf16(
                    false, ah, false, bl[ni], (short)0, acc[mi][ni], false, false);
            }
        }
    }
#pragma unroll
    for (int ni = 0; ni < 2; ++ni) {
        const int col = tileN0 + waveN * 32 + ni * 16 + l16;
        const float bcol = bias[col];
#pragma unroll
        for (int mi = 0; mi < 4; ++mi) {
            const int row0 = tileM0 + waveM * 64 + mi * 16 + half * 8;
#pragma unroll
            for (int r = 0; r < 8; ++r) {
                out[(size_t)(row0 + r) * N + col] = acc[mi][ni][r] + bcol;
            }
        }
    }
}

// ------------------------------------------------------------------ launch ---
extern "C" void kernel_launch(void* const* d_in, const int* in_sizes, int n_in,
                              void* d_out, int out_size, void* d_ws, size_t ws_size,
                              hipStream_t stream) {
    (void)n_in; (void)out_size;
    const float* X    = (const float*)d_in[0];   // (B*Xrows, C) fp32
    const float* W    = (const float*)d_in[1];   // (N, C) fp32
    const float* bias = (const float*)d_in[2];   // (N,)
    const float* SX   = (const float*)d_in[3];   // (B*Xrows, C/32)
    const float* SW   = (const float*)d_in[4];   // (N, C/32)
    float* out = (float*)d_out;                  // (B*Xrows, N)

    const int N = in_sizes[2];
    const int C = in_sizes[1] / N;
    const int M = (int)((long)in_sizes[0] / C);
    const int nblk = C / 32;

    dim3 grid((unsigned)(N / TN), (unsigned)(M / TM));
    const size_t need = ((size_t)M * C + (size_t)N * C) * 2 * sizeof(__bf16);

    if (ws_size >= need) {
        __bf16* aHi = (__bf16*)d_ws;
        __bf16* aLo = aHi + (size_t)M * C;
        __bf16* bHi = aLo + (size_t)M * C;
        __bf16* bLo = bHi + (size_t)N * C;

        long ta4 = (long)M * C / 4, tb4 = (long)N * C / 4;
        mx_split_kernel<<<(unsigned)((ta4 + 255) / 256), 256, 0, stream>>>(
            X, SX, aHi, aLo, C, nblk, ta4);
        mx_split_kernel<<<(unsigned)((tb4 + 255) / 256), 256, 0, stream>>>(
            W, SW, bHi, bLo, C, nblk, tb4);

        const int smemBytes = 2 * 4 * TM * LDT * (int)sizeof(__bf16);  // 80KB
        (void)hipFuncSetAttribute((const void*)mxfp_gemm_v2_kernel,
                                  hipFuncAttributeMaxDynamicSharedMemorySize,
                                  smemBytes);
        mxfp_gemm_v2_kernel<<<grid, 256, smemBytes, stream>>>(
            aHi, aLo, bHi, bLo, bias, out, M, N, C);
    } else {
        mxfp_gemm_fused_kernel<<<grid, 256, 0, stream>>>(
            X, W, bias, SX, SW, out, M, N, C);
    }
}
